// MultiHeadAttention_76922864272034
// MI455X (gfx1250) — matrix operand order repaired and checked by reading, not compiled, not run on hardware
//
#include <hip/hip_runtime.h>

#define DMODEL 1024
#define S_LEN  2048
#define BATCH  2
#define NHEAD  16
#define DK     64
#define M_TOT  (BATCH * S_LEN)     // 4096 rows
#define N_QKV  (3 * DMODEL)        // 3072
#define BH     (BATCH * NHEAD)     // 32

typedef __attribute__((ext_vector_type(16))) __bf16       v16bf;
typedef __attribute__((ext_vector_type(8)))  float        v8f;
typedef __attribute__((ext_vector_type(4)))  unsigned int v4u;

union Frag {
    v16bf v;
    v4u   q[2];
};

__device__ __forceinline__ v8f zero8() {
    v8f z = {0.f, 0.f, 0.f, 0.f, 0.f, 0.f, 0.f, 0.f};
    return z;
}

__device__ __forceinline__ unsigned short f2bf(float f) {
    union { float f; unsigned u; } cv; cv.f = f;
    unsigned u = cv.u;
    u += 0x7FFFu + ((u >> 16) & 1u);       // round-to-nearest-even
    return (unsigned short)(u >> 16);
}

__device__ __forceinline__ v8f bwmma(const Frag& a, const Frag& b, v8f c) {
    return __builtin_amdgcn_wmma_f32_16x16x32_bf16(
        /*neg_a=*/false, a.v, /*neg_b=*/false, b.v,
        /*c_mod=*/(short)0, c, /*reuse_a=*/false, /*reuse_b=*/false);
}

// A fragment: 16 rows x 32 K, row = lane&15; K halves split by half-wave (+hi8)
__device__ __forceinline__ Frag ldA(const unsigned short* row, int k0, int hi8) {
    Frag f;
    f.q[0] = *(const v4u*)(row + k0 + hi8);
    f.q[1] = *(const v4u*)(row + k0 + 16 + hi8);
    return f;
}
// B fragment: 8 bf16 at p and 8 bf16 at p+16 (K split by half-wave, +hi8, folded into p)
__device__ __forceinline__ Frag ldB(const unsigned short* p) {
    Frag f;
    f.q[0] = *(const v4u*)(p);
    f.q[1] = *(const v4u*)(p + 16);
    return f;
}

// --------------------------------------------------- shared GEMM chunk ops --
struct B4 { Frag b[4]; };

__device__ __forceinline__ void loadB4(B4& bb,
                                       const unsigned short* const* wrow, int k) {
#pragma unroll
    for (int j = 0; j < 4; ++j) bb.b[j] = ldB(wrow[j] + k);
}
__device__ __forceinline__ void loadA2(Frag& a0, Frag& a1,
                                       const unsigned short* arow0,
                                       const unsigned short* arow1,
                                       int k, int hi8) {
    a0 = ldA(arow0, k, hi8);
    a1 = ldA(arow1, k, hi8);
}
__device__ __forceinline__ void mma8(const Frag& a0, const Frag& a1,
                                     const B4& bb, v8f acc[2][4]) {
#pragma unroll
    for (int j = 0; j < 4; ++j) {
        acc[0][j] = bwmma(a0, bb.b[j], acc[0][j]);
        acc[1][j] = bwmma(a1, bb.b[j], acc[1][j]);
    }
}

// Fully-unrolled, software-pipelined K reduction over a 32x64 wave tile.
// Straight-line after unroll: no loop-carried PHIs -> no register shuttles.
__device__ __forceinline__ void gemmCore(v8f acc[2][4],
                                         const unsigned short* arow0,
                                         const unsigned short* arow1,
                                         const unsigned short* const* wrow,
                                         int hi8) {
    Frag a0, a1;
    B4 b0, b1;
    loadA2(a0, a1, arow0, arow1, 0, hi8);
    loadB4(b0, wrow, 0);
#pragma unroll
    for (int kc = 0; kc < (DMODEL - 64) / 64; ++kc) {   // 15 steady iterations
        const int k0 = kc * 64;
        loadB4(b1, wrow, k0 + 32);
        mma8(a0, a1, b0, acc);
        loadA2(a0, a1, arow0, arow1, k0 + 32, hi8);
        loadB4(b0, wrow, k0 + 64);
        mma8(a0, a1, b1, acc);
        loadA2(a0, a1, arow0, arow1, k0 + 64, hi8);
    }
    loadB4(b1, wrow, DMODEL - 32);
    mma8(a0, a1, b0, acc);
    loadA2(a0, a1, arow0, arow1, DMODEL - 32, hi8);
    mma8(a0, a1, b1, acc);
}

// ---------------------------------------------------------------- converts --
__global__ void cvt_bf16(const float* __restrict__ src,
                         unsigned short* __restrict__ dst, int n) {
    int i = blockIdx.x * blockDim.x + threadIdx.x;
    if (i < n) dst[i] = f2bf(src[i]);
}

// dst[n*rowsK + k] = src[k*colsN + n]  (weight transpose to [N][K])
__global__ void cvt_transpose(const float* __restrict__ src,
                              unsigned short* __restrict__ dst,
                              int rowsK, int colsN) {
    int i = blockIdx.x * blockDim.x + threadIdx.x;
    if (i >= rowsK * colsN) return;
    int n = i / rowsK;
    int k = i - n * rowsK;
    dst[i] = f2bf(src[(size_t)k * colsN + n]);
}

// ---------------------------------------------------------------- QKV GEMM --
// block = 128 threads (4 waves); block tile 32 x 256; wave tile 32 x 64.
__global__ __launch_bounds__(128) void qkv_gemm(
        const unsigned short* __restrict__ Xb,   // [4096][1024] bf16
        const unsigned short* __restrict__ Wt,   // [3072][1024] bf16 (W^T)
        const float* __restrict__ bias,          // [3072] f32
        unsigned short* __restrict__ Qb,         // [BH][S][DK], pre-scaled
        unsigned short* __restrict__ Kb,         // [BH][S][DK]
        unsigned short* __restrict__ Vt)         // [BH][DK][S]
{
    const int lane = threadIdx.x & 31;
    const int wave = threadIdx.x >> 5;
    const int m0   = blockIdx.x * 32;
    const int n0   = blockIdx.y * 256 + wave * 64;
    const int nl   = lane & 15;
    const int hi8  = (lane < 16) ? 0 : 8;
    const int hi16 = (lane < 16) ? 0 : 16;

    v8f acc[2][4];
#pragma unroll
    for (int mi = 0; mi < 2; ++mi)
#pragma unroll
        for (int j = 0; j < 4; ++j) acc[mi][j] = zero8();

    const unsigned short* arow0 = Xb + (size_t)(m0 + nl) * DMODEL;
    const unsigned short* arow1 = arow0 + (size_t)16 * DMODEL;
    const unsigned short* wrow[4];
#pragma unroll
    for (int j = 0; j < 4; ++j)
        wrow[j] = Wt + (size_t)(n0 + j * 16 + nl) * DMODEL + hi8;

    gemmCore(acc, arow0, arow1, wrow, hi8);

#pragma unroll
    for (int j = 0; j < 4; ++j) {
        const int n     = n0 + j * 16 + nl;
        const float bv  = bias[n];
        const int which = n >> 10;       // 0=Q 1=K 2=V
        const int d     = n & 1023;
        const int h     = d >> 6;
        const int dkk   = d & 63;
#pragma unroll
        for (int mi = 0; mi < 2; ++mi) {
#pragma unroll
            for (int r = 0; r < 8; ++r) {
                const int row  = m0 + mi * 16 + r + hi8;
                const int bidx = row >> 11;
                const int s    = row & 2047;
                const float val = acc[mi][j][r] + bv;
                const size_t sd =
                    ((size_t)(bidx * NHEAD + h) * S_LEN + s) * DK + dkk;
                if (which == 0)      Qb[sd] = f2bf(val * 0.125f); // 1/sqrt(64)
                else if (which == 1) Kb[sd] = f2bf(val);
                else Vt[((size_t)(bidx * NHEAD + h) * DK + dkk) * S_LEN + s] =
                         f2bf(val);
            }
        }
    }
}

// ---------------------------------------------------------- flash attention --
__device__ __forceinline__ void loadKtile(const unsigned short* kb0, int t,
                                          int nl, int hi16, Frag kf[4]) {
    const unsigned short* p0 = kb0 + (size_t)(t + nl) * DK + hi16 / 2;
    const unsigned short* p1 = kb0 + (size_t)(t + 16 + nl) * DK + hi16 / 2;
    kf[0] = ldB(p0); kf[1] = ldB(p0 + 32);
    kf[2] = ldB(p1); kf[3] = ldB(p1 + 32);
}
__device__ __forceinline__ void loadVtile(const unsigned short* vb0, int t,
                                          int nl, int hi16, Frag vf[4]) {
    const unsigned short* vb = vb0 + t + hi16 / 2;
    vf[0] = ldB(vb + (size_t)(nl)      * S_LEN);
    vf[1] = ldB(vb + (size_t)(16 + nl) * S_LEN);
    vf[2] = ldB(vb + (size_t)(32 + nl) * S_LEN);
    vf[3] = ldB(vb + (size_t)(48 + nl) * S_LEN);
}

// One 32-key tile. V for *this* tile loads at the top (consumed after the
// score WMMAs + K prefetch + softmax chain); K for the *next* tile prefetches
// into kfn right after the score WMMAs (hidden under the softmax).
__device__ __forceinline__ void attnTile(
        const Frag& qa0, const Frag& qa1,
        const Frag kf[4], Frag kfn[4],
        const unsigned short* kb0, const unsigned short* vb0,
        int t, int tnext,
        v8f& o0, v8f& o1, v8f& o2, v8f& o3,
        float rmax[8], float rsum[8],
        unsigned short* myP, int nl, int hi8, int hi16)
{
    // V tile for this iteration (single-buffered; long distance to use)
    Frag vf[4];
    loadVtile(vb0, t, nl, hi16, vf);

    // ---- scores: S[16x32] = Q @ K^T (scale folded into Q) ----
    v8f s0 = zero8(), s1 = zero8();
    s0 = bwmma(qa0, kf[0], s0);
    s0 = bwmma(qa1, kf[1], s0);
    s1 = bwmma(qa0, kf[2], s1);
    s1 = bwmma(qa1, kf[3], s1);

    // prefetch next K tile
    loadKtile(kb0, tnext, nl, hi16, kfn);

    // ---- online softmax (half-wave row reductions) ----
    float corr[8];
#pragma unroll
    for (int r = 0; r < 8; ++r) {
        float mx = fmaxf(s0[r], s1[r]);
        mx = fmaxf(mx, __shfl_xor(mx, 1, 32));
        mx = fmaxf(mx, __shfl_xor(mx, 2, 32));
        mx = fmaxf(mx, __shfl_xor(mx, 4, 32));
        mx = fmaxf(mx, __shfl_xor(mx, 8, 32));
        const float nm = fmaxf(rmax[r], mx);
        const float c  = __expf(rmax[r] - nm);
        rmax[r] = nm;
        const float p0 = __expf(s0[r] - nm);
        const float p1 = __expf(s1[r] - nm);
        float ls = p0 + p1;
        ls += __shfl_xor(ls, 1, 32);
        ls += __shfl_xor(ls, 2, 32);
        ls += __shfl_xor(ls, 4, 32);
        ls += __shfl_xor(ls, 8, 32);
        rsum[r] = rsum[r] * c + ls;
        corr[r] = c;
        const int row = r + hi8;                 // C-layout row of this lane
        myP[row * 32 + nl]      = f2bf(p0);
        myP[row * 32 + 16 + nl] = f2bf(p1);
    }
#pragma unroll
    for (int r = 0; r < 8; ++r) {
        o0[r] *= corr[r]; o1[r] *= corr[r];
        o2[r] *= corr[r]; o3[r] *= corr[r];
    }

    // ---- reshape P (C-layout in LDS) into A-fragment layout ----
    Frag pa;
    {
        const unsigned short* pr = myP + nl * 32 + hi8;
        pa.q[0] = *(const v4u*)(pr);
        pa.q[1] = *(const v4u*)(pr + 16);
    }

    // ---- O += P @ V ----
    o0 = bwmma(pa, vf[0], o0);
    o1 = bwmma(pa, vf[1], o1);
    o2 = bwmma(pa, vf[2], o2);
    o3 = bwmma(pa, vf[3], o3);
}

// grid (S/64, BH); 4 waves/block, one 16-row query tile per wave.
// 2x-unrolled key loop with ping-pong K fragment buffers.
__global__ __launch_bounds__(128) void flash_attn(
        const unsigned short* __restrict__ Qb,   // [BH][S][DK] pre-scaled
        const unsigned short* __restrict__ Kb,   // [BH][S][DK]
        const unsigned short* __restrict__ Vt,   // [BH][DK][S]
        unsigned short* __restrict__ Cb)         // [B*S][DMODEL] bf16 context
{
    __shared__ __align__(16) unsigned short pS[4][16 * 32];

    const int lane = threadIdx.x & 31;
    const int wave = threadIdx.x >> 5;
    const int bh   = blockIdx.y;
    const int q0   = blockIdx.x * 64 + wave * 16;
    const int nl   = lane & 15;
    const int hi   = (lane < 16) ? 0 : 1;
    const int hi8  = hi * 8;
    const int hi16 = hi * 16;

    // Q fragments: 16x64 as two 16x32 A operands (loaded once)
    const unsigned short* qrow = Qb + ((size_t)bh * S_LEN + q0 + nl) * DK;
    Frag qa0, qa1;
    qa0.q[0] = *(const v4u*)(qrow + 0  + hi8);
    qa0.q[1] = *(const v4u*)(qrow + 16 + hi8);
    qa1.q[0] = *(const v4u*)(qrow + 32 + hi8);
    qa1.q[1] = *(const v4u*)(qrow + 48 + hi8);

    v8f o0 = zero8(), o1 = zero8(), o2 = zero8(), o3 = zero8();
    float rmax[8], rsum[8];
#pragma unroll
    for (int r = 0; r < 8; ++r) { rmax[r] = -1e30f; rsum[r] = 0.f; }

    unsigned short* myP = pS[wave];
    const unsigned short* kb0 = Kb + (size_t)bh * S_LEN * DK;
    const unsigned short* vb0 = Vt + (size_t)bh * DK * S_LEN;

    Frag kfA[4], kfB[4];
    loadKtile(kb0, 0, nl, hi16, kfA);

    for (int t0 = 0; t0 < S_LEN; t0 += 64) {
        const int t1 = t0 + 32;                              // always in-bounds
        const int t2 = (t0 + 64 < S_LEN) ? (t0 + 64) : t1;   // clamped
        // tile t0: compute from A buffer, prefetch t1 into B buffer
        attnTile(qa0, qa1, kfA, kfB, kb0, vb0, t0, t1,
                 o0, o1, o2, o3, rmax, rsum, myP, nl, hi8, hi16);
        // tile t1: compute from B buffer, prefetch t2 into A buffer
        attnTile(qa0, qa1, kfB, kfA, kb0, vb0, t1, t2,
                 o0, o1, o2, o3, rmax, rsum, myP, nl, hi8, hi16);
    }

    // ---- normalize + write context (bf16, row-major [B*S][DMODEL]) ----
    const int h = bh & (NHEAD - 1);
    const int b = bh >> 4;
#pragma unroll
    for (int r = 0; r < 8; ++r) {
        const float inv = 1.0f / rsum[r];
        const int row   = q0 + r + hi8;
        const size_t bs = ((size_t)(b * S_LEN + row)) * DMODEL + h * DK + nl;
        Cb[bs + 0]  = f2bf(o0[r] * inv);
        Cb[bs + 16] = f2bf(o1[r] * inv);
        Cb[bs + 32] = f2bf(o2[r] * inv);
        Cb[bs + 48] = f2bf(o3[r] * inv);
    }
}

// -------------------------------------------------------------- output GEMM --
__global__ __launch_bounds__(128) void out_gemm(
        const unsigned short* __restrict__ Cb,   // [4096][1024] bf16
        const unsigned short* __restrict__ Wt,   // [1024][1024] bf16 (W_out^T)
        float* __restrict__ out)                 // [4096][1024] f32
{
    const int lane = threadIdx.x & 31;
    const int wave = threadIdx.x >> 5;
    const int m0   = blockIdx.x * 32;
    const int n0   = blockIdx.y * 256 + wave * 64;
    const int nl   = lane & 15;
    const int hi8  = (lane < 16) ? 0 : 8;
    const int hi16 = (lane < 16) ? 0 : 16;

    v8f acc[2][4];
#pragma unroll
    for (int mi = 0; mi < 2; ++mi)
#pragma unroll
        for (int j = 0; j < 4; ++j) acc[mi][j] = zero8();

    const unsigned short* arow0 = Cb + (size_t)(m0 + nl) * DMODEL;
    const unsigned short* arow1 = arow0 + (size_t)16 * DMODEL;
    const unsigned short* wrow[4];
#pragma unroll
    for (int j = 0; j < 4; ++j)
        wrow[j] = Wt + (size_t)(n0 + j * 16 + nl) * DMODEL + hi8;

    gemmCore(acc, arow0, arow1, wrow, hi8);

#pragma unroll
    for (int j = 0; j < 4; ++j) {
        const int n = n0 + j * 16 + nl;
#pragma unroll
        for (int mi = 0; mi < 2; ++mi) {
#pragma unroll
            for (int r = 0; r < 8; ++r) {
                const int row = m0 + mi * 16 + r + hi8;
                out[(size_t)row * DMODEL + n] = acc[mi][j][r];
            }
        }
    }
}

// ------------------------------------------------------------------ launch --
extern "C" void kernel_launch(void* const* d_in, const int* in_sizes, int n_in,
                              void* d_out, int out_size, void* d_ws, size_t ws_size,
                              hipStream_t stream) {
    const float* X    = (const float*)d_in[0];   // [2,2048,1024]
    const float* Wqkv = (const float*)d_in[1];   // [1024,3072]
    const float* bqkv = (const float*)d_in[2];   // [3072]
    const float* Wout = (const float*)d_in[3];   // [1024,1024]
    float* out = (float*)d_out;

    char* ws = (char*)d_ws;
    size_t off = 0;
    unsigned short* Xb  = (unsigned short*)(ws + off); off += (size_t)M_TOT * DMODEL * 2;
    unsigned short* Wqt = (unsigned short*)(ws + off); off += (size_t)N_QKV * DMODEL * 2;
    unsigned short* Wot = (unsigned short*)(ws + off); off += (size_t)DMODEL * DMODEL * 2;
    unsigned short* Qb  = (unsigned short*)(ws + off); off += (size_t)BH * S_LEN * DK * 2;
    unsigned short* Kb  = (unsigned short*)(ws + off); off += (size_t)BH * S_LEN * DK * 2;
    unsigned short* Vt  = (unsigned short*)(ws + off); off += (size_t)BH * S_LEN * DK * 2;
    unsigned short* Cb  = (unsigned short*)(ws + off); off += (size_t)M_TOT * DMODEL * 2;

    {
        int n = M_TOT * DMODEL;
        cvt_bf16<<<(n + 255) / 256, 256, 0, stream>>>(X, Xb, n);
    }
    {
        int n = N_QKV * DMODEL;
        cvt_transpose<<<(n + 255) / 256, 256, 0, stream>>>(Wqkv, Wqt, DMODEL, N_QKV);
    }
    {
        int n = DMODEL * DMODEL;
        cvt_transpose<<<(n + 255) / 256, 256, 0, stream>>>(Wout, Wot, DMODEL, DMODEL);
    }
    qkv_gemm<<<dim3(M_TOT / 32, N_QKV / 256), 128, 0, stream>>>(Xb, Wqt, bqkv, Qb, Kb, Vt);
    flash_attn<<<dim3(S_LEN / 64, BH), 128, 0, stream>>>(Qb, Kb, Vt, Cb);
    out_gemm<<<dim3(M_TOT / 32, DMODEL / 256), 128, 0, stream>>>(Cb, Wot, out);
}
